// SpearmanCorrelationLossOptimized_43069932045051
// MI455X (gfx1250) — compile-verified, matrix operand order checked by reading, not run on hardware
//
#include <hip/hip_runtime.h>
#include <hip/hip_bf16.h>

typedef _Float16 half8 __attribute__((ext_vector_type(8)));
typedef _Float16 v16h  __attribute__((ext_vector_type(16)));
typedef float    v8f   __attribute__((ext_vector_type(8)));

union H16 { v16h v; half8 h[2]; };

// ---------------------------------------------------------------- zero output
__global__ __launch_bounds__(64) void spcl_zero_out(float* out) {
    if (threadIdx.x == 0 && blockIdx.x == 0) out[0] = 0.0f;
}

// ------------------------------------------------- row L2-normalize + f16 cast
__global__ __launch_bounds__(256) void spcl_normalize_f16(
        const float* __restrict__ R, _Float16* __restrict__ Nh, int D) {
    const int row = blockIdx.x;
    const float* r = R + (size_t)row * D;
    __shared__ float red[256];
    float ss = 0.0f;
    for (int d = threadIdx.x; d < D; d += 256) { float v = r[d]; ss += v * v; }
    red[threadIdx.x] = ss;
    __syncthreads();
    for (int s = 128; s > 0; s >>= 1) {
        if (threadIdx.x < s) red[threadIdx.x] += red[threadIdx.x + s];
        __syncthreads();
    }
    const float inv = 1.0f / fmaxf(sqrtf(red[0]), 1e-12f);
    _Float16* o = Nh + (size_t)row * D;
    for (int d = threadIdx.x; d < D; d += 256) o[d] = (_Float16)(r[d] * inv);
}

// ---------------------------------------------------- S = N * N^T  via WMMA f16
// One wave computes a 16(M) x 64(N) strip: A-frag reused against 4 B-frags.
// A and B fragments load with the identical pattern since B cols == N rows.
__global__ __launch_bounds__(256) void spcl_gemm_wmma(
        const _Float16* __restrict__ Nh, float* __restrict__ S, int B, int D) {
    const int lane  = threadIdx.x & 31;
    const int wid   = blockIdx.x * (blockDim.x >> 5) + (threadIdx.x >> 5);
    const int ngrps = B >> 6;                 // column groups of 64
    const int mt    = wid / ngrps;
    if (mt >= (B >> 4)) return;               // uniform per wave
    const int m0 = mt * 16;
    const int n0 = (wid % ngrps) * 64;

    const int rsel = lane & 15;               // row within 16-row tile
    const int koff = (lane >> 4) * 8;         // lanes 16-31: K base +8

    const _Float16* arow  = Nh + (size_t)(m0      + rsel) * D;
    const _Float16* brow0 = Nh + (size_t)(n0      + rsel) * D;
    const _Float16* brow1 = Nh + (size_t)(n0 + 16 + rsel) * D;
    const _Float16* brow2 = Nh + (size_t)(n0 + 32 + rsel) * D;
    const _Float16* brow3 = Nh + (size_t)(n0 + 48 + rsel) * D;

    v8f c0 = {}, c1 = {}, c2 = {}, c3 = {};
    for (int k0 = 0; k0 < D; k0 += 32) {
        H16 a, b0, b1, b2, b3;
        a.h[0]  = *(const half8*)(arow  + k0 + koff);
        a.h[1]  = *(const half8*)(arow  + k0 + 16 + koff);
        b0.h[0] = *(const half8*)(brow0 + k0 + koff);
        b0.h[1] = *(const half8*)(brow0 + k0 + 16 + koff);
        b1.h[0] = *(const half8*)(brow1 + k0 + koff);
        b1.h[1] = *(const half8*)(brow1 + k0 + 16 + koff);
        b2.h[0] = *(const half8*)(brow2 + k0 + koff);
        b2.h[1] = *(const half8*)(brow2 + k0 + 16 + koff);
        b3.h[0] = *(const half8*)(brow3 + k0 + koff);
        b3.h[1] = *(const half8*)(brow3 + k0 + 16 + koff);
        c0 = __builtin_amdgcn_wmma_f32_16x16x32_f16(false, a.v, false, b0.v,
                                                    (short)0, c0, false, false);
        c1 = __builtin_amdgcn_wmma_f32_16x16x32_f16(false, a.v, false, b1.v,
                                                    (short)0, c1, false, false);
        c2 = __builtin_amdgcn_wmma_f32_16x16x32_f16(false, a.v, false, b2.v,
                                                    (short)0, c2, false, false);
        c3 = __builtin_amdgcn_wmma_f32_16x16x32_f16(false, a.v, false, b3.v,
                                                    (short)0, c3, false, false);
    }
    // C/D layout: VGPR r -> M = r (lanes 0-15) or r+8 (lanes 16-31); N = lane&15
    const int col   = lane & 15;
    const int rbase = (lane >> 4) * 8;
#pragma unroll
    for (int r = 0; r < 8; ++r) {
        const size_t rowoff = (size_t)(m0 + rbase + r) * B;
        S[rowoff + n0 +      col] = c0[r];
        S[rowoff + n0 + 16 + col] = c1[r];
        S[rowoff + n0 + 32 + col] = c2[r];
        S[rowoff + n0 + 48 + col] = c3[r];
    }
}

// ------------------------------- per-row: soft_rank (x2) + Pearson corr + accum
// shared layout: sv[n] f32 | si[n] i32 | pm[n] f32 | pw[n] f32 | ra[n] | rb[n]
__global__ __launch_bounds__(256) void spcl_rank_corr(
        const float* __restrict__ S, const float* __restrict__ L,
        float* __restrict__ out, int n) {
    const int tid = threadIdx.x;
    const int row = blockIdx.x;
    extern __shared__ unsigned char smem[];
    float* sv = (float*)smem;
    int*   si = (int*)(sv + n);
    float* pm = (float*)(si + n);
    float* pw = pm + n;
    float* ra = pw + n;
    float* rb = ra + n;
    __shared__ int sh_t;

    const float* srow = S + (size_t)row * n;
    const float  lrow = L[row];

    for (int pass = 0; pass < 2; ++pass) {
        float* tgt = (pass == 0) ? ra : rb;
        // theta = values / REG_STRENGTH (0.1)
        for (int j = tid; j < n; j += 256) {
            float v = (pass == 0) ? (-srow[j]) : fabsf(lrow - L[j]);
            sv[j] = v * 10.0f;
            si[j] = j;
        }
        __syncthreads();
        // bitonic sort, descending (argsort(-theta))
        for (int k = 2; k <= n; k <<= 1) {
            for (int j = k >> 1; j > 0; j >>= 1) {
                for (int i = tid; i < n; i += 256) {
                    int ixj = i ^ j;
                    if (ixj > i) {
                        float a = sv[i], b = sv[ixj];
                        bool descSeg = ((i & k) == 0);
                        if (descSeg ? (a < b) : (a > b)) {
                            sv[i] = b; sv[ixj] = a;
                            int t = si[i]; si[i] = si[ixj]; si[ixj] = t;
                        }
                    }
                }
                __syncthreads();
            }
        }
        // PAVA (non-increasing isotonic regression of s - [n..1]), thread 0
        if (tid == 0) {
            int t = 0;
            for (int j = 0; j < n; ++j) {
                float y = sv[j] - (float)(n - j);
                pm[t] = y; pw[t] = 1.0f; ++t;
                while (t > 1 && pm[t - 2] <= pm[t - 1]) {
                    float nw = pw[t - 2] + pw[t - 1];
                    pm[t - 2] = (pm[t - 2] * pw[t - 2] + pm[t - 1] * pw[t - 1]) / nw;
                    pw[t - 2] = nw;
                    --t;
                }
            }
            for (int p = 1; p < t; ++p) pw[p] += pw[p - 1];   // cumw
            sh_t = t;
        }
        __syncthreads();
        const int t = sh_t;
        // dual[k] = pool mean, via searchsorted(cumw, k+0.5); scatter inverse perm
        for (int k = tid; k < n; k += 256) {
            float pos = (float)k + 0.5f;
            int lo = 0, hi = t - 1;
            while (lo < hi) {
                int mid = (lo + hi) >> 1;
                if (pw[mid] > pos) hi = mid; else lo = mid + 1;
            }
            tgt[si[k]] = sv[k] - pm[lo];       // ranks[perm[k]] = s[k]-dual[k]
        }
        __syncthreads();
    }

    // Pearson correlation of ra vs rb  (pm/pw free now; reuse as reduction bufs)
    float sa = 0.0f, sb = 0.0f;
    for (int j = tid; j < n; j += 256) { sa += ra[j]; sb += rb[j]; }
    pm[tid] = sa; pw[tid] = sb;
    __syncthreads();
    for (int s = 128; s > 0; s >>= 1) {
        if (tid < s) { pm[tid] += pm[tid + s]; pw[tid] += pw[tid + s]; }
        __syncthreads();
    }
    const float ma = pm[0] / (float)n;
    const float mb = pw[0] / (float)n;
    __syncthreads();
    float cab = 0.0f, caa = 0.0f, cbb = 0.0f;
    for (int j = tid; j < n; j += 256) {
        float x = ra[j] - ma, y = rb[j] - mb;
        cab += x * y; caa += x * x; cbb += y * y;
    }
    __syncthreads();                 // done reading ra before reusing it
    pm[tid] = cab; pw[tid] = caa; ra[tid] = cbb;
    __syncthreads();
    for (int s = 128; s > 0; s >>= 1) {
        if (tid < s) { pm[tid] += pm[tid + s]; pw[tid] += pw[tid + s]; ra[tid] += ra[tid + s]; }
        __syncthreads();
    }
    if (tid == 0) {
        float corr = pm[0] / (sqrtf(pw[0]) * sqrtf(ra[0]) + 1e-8f);
        atomicAdd(out, -corr / (float)n);
    }
}

extern "C" void kernel_launch(void* const* d_in, const int* in_sizes, int n_in,
                              void* d_out, int out_size, void* d_ws, size_t ws_size,
                              hipStream_t stream) {
    (void)n_in; (void)out_size;
    const float* reps   = (const float*)d_in[0];
    const float* labels = (const float*)d_in[1];
    const int B = in_sizes[1];                 // 2048
    const int D = in_sizes[0] / B;             // 512

    const size_t nh_bytes = (size_t)B * D * sizeof(_Float16);
    const size_t s_bytes  = (size_t)B * B * sizeof(float);
    if (ws_size < nh_bytes + s_bytes) return;  // workspace too small: no-op

    _Float16* Nh = (_Float16*)d_ws;
    float*    S  = (float*)((char*)d_ws + nh_bytes);
    float*    out = (float*)d_out;

    spcl_zero_out<<<1, 64, 0, stream>>>(out);
    spcl_normalize_f16<<<B, 256, 0, stream>>>(reps, Nh, D);

    const int waves  = (B / 16) * (B / 64);    // one wave per 16x64 strip
    const int blocks = (waves + 7) / 8;        // 8 waves / 256-thread block
    spcl_gemm_wmma<<<blocks, 256, 0, stream>>>(Nh, S, B, D);

    const size_t shmem = (size_t)6 * B * sizeof(float);  // 48 KB for B=2048
    spcl_rank_corr<<<B, 256, shmem, stream>>>(S, labels, out, B);
}
